// Latent_83751862272681
// MI455X (gfx1250) — matrix operand order repaired and checked by reading, not compiled, not run on hardware
//
#include <hip/hip_runtime.h>

// ---------------------------------------------------------------------------
// Problem constants (match setup_inputs): B=32, z_dim=64, H=W=64, M=512
// ---------------------------------------------------------------------------
#define N_PTS    131072   // B*H*W
#define ZDIM     64
#define MGROUP   512      // group size (reduction axis of logsumexp)
#define NGROUPS  256      // N_PTS / MGROUP
#define NCODES   512      // codebook entries (output columns of lse)
#define CHUNK_M  128      // rows of z staged in LDS per iteration
#define NCHUNK   (MGROUP / CHUNK_M)

typedef __attribute__((ext_vector_type(16))) __bf16 bf16x16;
typedef __attribute__((ext_vector_type(8)))  __bf16 bf16x8;
typedef __attribute__((ext_vector_type(8)))  float  f32x8;

static __device__ __forceinline__ bf16x16 cat8(bf16x8 lo, bf16x8 hi) {
  return __builtin_shufflevector(lo, hi, 0,1,2,3,4,5,6,7,8,9,10,11,12,13,14,15);
}

// ---------------------------------------------------------------------------
// Workspace layout (bytes)
// ---------------------------------------------------------------------------
#define WS_ZB     0u                                    // N_PTS*ZDIM bf16  (16 MB)
#define WS_EB     (N_PTS*ZDIM*2u)                       // NCODES*ZDIM bf16 (64 KB)
#define WS_ENORM  (WS_EB + NCODES*ZDIM*2u)              // NCODES f32
#define WS_ZNORM  (WS_ENORM + NCODES*4u)                // N_PTS f32
#define WS_ACCUM  (WS_ZNORM + N_PTS*4u)                 // 1 f32

// ---------------------------------------------------------------------------
// Async memory->LDS staging (CDNA5 GLOBAL_LOAD_ASYNC_TO_LDS_B128).
// Per CDNA5 ISA: INST_OFFSET is added to BOTH the global and the
// LDS address, so one (lds,vaddr) pair + 4 offsets covers a 16KB chunk with
// 256 threads x 16B x 4.
// ---------------------------------------------------------------------------
static __device__ __forceinline__ void async_copy_chunk16k(uint32_t lds_dst,
                                                           const __bf16* gsrc,
                                                           int tid) {
  uint32_t la = lds_dst + (uint32_t)tid * 16u;
  uint64_t ga = (uint64_t)(uintptr_t)gsrc + (uint32_t)tid * 16u;
  asm volatile(
      "global_load_async_to_lds_b128 %0, %1, off\n\t"
      "global_load_async_to_lds_b128 %0, %1, off offset:4096\n\t"
      "global_load_async_to_lds_b128 %0, %1, off offset:8192\n\t"
      "global_load_async_to_lds_b128 %0, %1, off offset:12288"
      :: "v"(la), "v"(ga) : "memory");
}

// ---------------------------------------------------------------------------
// Pre-pass: codebook -> bf16 + row norms (f32)
// ---------------------------------------------------------------------------
__global__ void prep_e_kernel(const float* __restrict__ e,
                              __bf16* __restrict__ eb,
                              float* __restrict__ enorm) {
  int r = blockIdx.x * blockDim.x + threadIdx.x;   // 512 rows
  if (r >= NCODES) return;
  float s = 0.f;
  #pragma unroll 8
  for (int d = 0; d < ZDIM; ++d) {
    float v = e[r * ZDIM + d];
    eb[r * ZDIM + d] = (__bf16)v;
    s += v * v;
  }
  enorm[r] = s;
}

// ---------------------------------------------------------------------------
// Pre-pass: zf = transpose(z,(0,2,3,1)).reshape(-1,D)[perm] -> bf16
// one thread per (n,d); writes coalesced; reads serviced by 192MB L2
// ---------------------------------------------------------------------------
__global__ void gather_z_kernel(const float* __restrict__ z,
                                const int* __restrict__ perm,
                                __bf16* __restrict__ zb) {
  int tid = blockIdx.x * blockDim.x + threadIdx.x;  // N_PTS*ZDIM threads
  int n = tid >> 6;
  int d = tid & 63;
  int np = perm[n];                 // flat (b,h,w) index
  int w = np & 63;
  int h = (np >> 6) & 63;
  int b = np >> 12;
  // z layout (B, z_dim, H, W)
  float v = z[(size_t)b * (ZDIM * 64 * 64) + (size_t)d * (64 * 64) + h * 64 + w];
  zb[tid] = (__bf16)v;
}

// ---------------------------------------------------------------------------
// Pre-pass: per-point squared norms (f32 accumulate over bf16 data)
// ---------------------------------------------------------------------------
__global__ void znorm_kernel(const __bf16* __restrict__ zb,
                             float* __restrict__ znorm) {
  int n = blockIdx.x * blockDim.x + threadIdx.x;    // N_PTS threads
  const bf16x8* p = (const bf16x8*)(zb + (size_t)n * ZDIM);
  float s = 0.f;
  #pragma unroll
  for (int j = 0; j < ZDIM / 8; ++j) {
    bf16x8 v = p[j];
    #pragma unroll
    for (int i = 0; i < 8; ++i) { float f = (float)v[i]; s += f * f; }
  }
  znorm[n] = s;
}

__global__ void init_accum_kernel(float* accum) { accum[0] = 0.f; }

// ---------------------------------------------------------------------------
// Main kernel: one workgroup per group g.
//   logits[m,e] = alpha*(||z_m||^2 + ||e_e||^2) - 2*alpha*(z_m . e_e)
//   streaming logsumexp over m (rows) per codebook column e.
// 8 waves; wave w owns codebook columns [w*64, w*64+64), B-frags in registers.
// z chunks double-buffered in LDS via async loads; chunk c+1 load overlaps
// chunk c WMMA compute.
// ---------------------------------------------------------------------------
__global__ __launch_bounds__(256) void vq_lse_kernel(
    const __bf16* __restrict__ zb, const __bf16* __restrict__ eb,
    const float* __restrict__ znorm, const float* __restrict__ enorm,
    const float* __restrict__ log_sigma, float* __restrict__ accum) {
  __shared__ __align__(16) __bf16 Az[2][CHUNK_M * ZDIM];  // 2 x 16 KB
  __shared__ float red[8];

  const int g    = blockIdx.x;
  const int tid  = threadIdx.x;
  const int wv   = tid >> 5;       // wave id 0..7 (wave32)
  const int lane = tid & 31;
  const int nloc = lane & 15;      // column-in-tile (C/D layout) & row (A layout)
  const int half = lane >> 4;      // lane half selects K/M sub-ranges

  const float ls    = log_sigma[0];
  const float alpha = -0.5f * __expf(-2.f * ls);   // -1/(2*exp(ls)^2), T = 1
  const float m2a   = -2.f * alpha;

  // --- B fragments: wave's 64 codebook columns, 4 tiles x 2 k-steps ---------
  // 16-bit B (32x16) layout: lane half h selects K base 8h;
  // per lane: K in [8h, 8h+8) then [16+8h, 24+8h) from codebook row ncol.
  bf16x16 Bf[4][2];
  float   ceterm[4];
  const int ecol0 = wv * 64;
  #pragma unroll
  for (int t = 0; t < 4; ++t) {
    const int ncol = ecol0 + t * 16 + nloc;
    ceterm[t] = alpha * enorm[ncol];
    const __bf16* brow = eb + (size_t)ncol * ZDIM;
    #pragma unroll
    for (int kh = 0; kh < 2; ++kh) {
      const __bf16* p = brow + kh * 32 + half * 8;
      bf16x8 lo = *(const bf16x8*)(p);
      bf16x8 hi = *(const bf16x8*)(p + 16);
      Bf[t][kh] = cat8(lo, hi);
    }
  }

  // --- streaming logsumexp state per owned column (register resident) -------
  float rmax[4], rsum[4];
  #pragma unroll
  for (int t = 0; t < 4; ++t) { rmax[t] = -__builtin_inff(); rsum[t] = 0.f; }

  const uint32_t lds_buf[2] = { (uint32_t)(uintptr_t)(&Az[0][0]),
                                (uint32_t)(uintptr_t)(&Az[1][0]) };
  const __bf16* gbase = zb + (size_t)g * MGROUP * ZDIM;

  // prologue: kick off chunk 0
  async_copy_chunk16k(lds_buf[0], gbase, tid);

  for (int c = 0; c < NCHUNK; ++c) {
    __syncthreads();   // all waves done computing with buf[(c+1)&1] (iter c-1)
    if (c + 1 < NCHUNK) {
      async_copy_chunk16k(lds_buf[(c + 1) & 1],
                          gbase + (size_t)(c + 1) * CHUNK_M * ZDIM, tid);
      asm volatile("s_wait_asynccnt 0x4" ::: "memory");  // chunk c done (in-order)
    } else {
      asm volatile("s_wait_asynccnt 0x0" ::: "memory");
    }
    __syncthreads();   // chunk c visible to every wave
    const __bf16* Ab = &Az[c & 1][0];

    for (int mt = 0; mt < CHUNK_M / 16; ++mt) {
      // 16-bit A (16x32) layout: lane row M = nloc, K base = half*8;
      // per lane: K in [kb, kb+8) then [kb+16, kb+24), per k-step.
      const __bf16* ar = Ab + (mt * 16 + nloc) * ZDIM + half * 8;
      bf16x16 a0, a1;
      {
        bf16x8 lo = *(const bf16x8*)(ar);
        bf16x8 hi = *(const bf16x8*)(ar + 16);
        a0 = cat8(lo, hi);
        lo = *(const bf16x8*)(ar + 32);
        hi = *(const bf16x8*)(ar + 48);
        a1 = cat8(lo, hi);
      }
      // row-norm terms: 8 consecutive f32, L2-resident
      const float* znp = znorm + g * MGROUP + c * CHUNK_M + mt * 16 + half * 8;
      const float4 zv0 = *(const float4*)(znp);
      const float4 zv1 = *(const float4*)(znp + 4);
      float zn[8] = { alpha * zv0.x, alpha * zv0.y, alpha * zv0.z, alpha * zv0.w,
                      alpha * zv1.x, alpha * zv1.y, alpha * zv1.z, alpha * zv1.w };

      #pragma unroll
      for (int t = 0; t < 4; ++t) {
        f32x8 acc = {0.f, 0.f, 0.f, 0.f, 0.f, 0.f, 0.f, 0.f};
        acc = __builtin_amdgcn_wmma_f32_16x16x32_bf16(false, a0, false, Bf[t][0],
                                                      (short)0, acc, false, false);
        acc = __builtin_amdgcn_wmma_f32_16x16x32_bf16(false, a1, false, Bf[t][1],
                                                      (short)0, acc, false, false);
        // C/D layout: element i -> row (half*8 + i), column nloc
        float l[8];
        float tm = -__builtin_inff();
        #pragma unroll
        for (int i = 0; i < 8; ++i) {
          l[i] = fmaf(m2a, acc[i], zn[i] + ceterm[t]);
          tm = fmaxf(tm, l[i]);
        }
        tm = fmaxf(tm, __shfl_xor(tm, 16));          // combine 16-row tile halves
        const float nm = fmaxf(rmax[t], tm);
        float s = 0.f;
        #pragma unroll
        for (int i = 0; i < 8; ++i) s += __expf(l[i] - nm);
        s += __shfl_xor(s, 16);
        rsum[t] = rsum[t] * __expf(rmax[t] - nm) + s;
        rmax[t] = nm;
      }
    }
  }

  // --- reduce: sum of lse over this wave's 64 columns -----------------------
  float lsum = 0.f;
  #pragma unroll
  for (int t = 0; t < 4; ++t) lsum += rmax[t] + __logf(rsum[t]);
  if (half) lsum = 0.f;                 // halves hold duplicate column state
  #pragma unroll
  for (int off = 16; off >= 1; off >>= 1) lsum += __shfl_xor(lsum, off);
  if (lane == 0) red[wv] = lsum;
  __syncthreads();
  if (tid == 0) {
    float bs = 0.f;
    #pragma unroll
    for (int w = 0; w < 8; ++w) bs += red[w];
    atomicAdd(accum, bs);
  }
}

// ---------------------------------------------------------------------------
// Finalize: mean over (g, e), add constants.
// loss = -sum(lse)/(G*E) + 0.5*D*(2*ls - 1) + log(M)
// ---------------------------------------------------------------------------
__global__ void finalize_kernel(const float* __restrict__ accum,
                                const float* __restrict__ log_sigma,
                                float* __restrict__ out) {
  const float ls = log_sigma[0];
  out[0] = -(accum[0] / (float)(NGROUPS * NCODES))
           + 0.5f * (float)ZDIM * (2.f * ls - 1.f)
           + __logf((float)NCODES);
}

// ---------------------------------------------------------------------------
extern "C" void kernel_launch(void* const* d_in, const int* in_sizes, int n_in,
                              void* d_out, int out_size, void* d_ws, size_t ws_size,
                              hipStream_t stream) {
  const float* z         = (const float*)d_in[0];
  const float* e         = (const float*)d_in[1];
  const float* log_sigma = (const float*)d_in[2];
  const int*   perm      = (const int*)d_in[3];
  float* out = (float*)d_out;

  char* ws = (char*)d_ws;
  __bf16* zb     = (__bf16*)(ws + WS_ZB);
  __bf16* eb     = (__bf16*)(ws + WS_EB);
  float*  enormf = (float*)(ws + WS_ENORM);
  float*  znormf = (float*)(ws + WS_ZNORM);
  float*  accum  = (float*)(ws + WS_ACCUM);

  init_accum_kernel<<<1, 1, 0, stream>>>(accum);
  prep_e_kernel<<<(NCODES + 255) / 256, 256, 0, stream>>>(e, eb, enormf);
  gather_z_kernel<<<(N_PTS * ZDIM) / 256, 256, 0, stream>>>(z, perm, zb);
  znorm_kernel<<<N_PTS / 256, 256, 0, stream>>>(zb, znormf);
  vq_lse_kernel<<<NGROUPS, 256, 0, stream>>>(zb, eb, znormf, enormf, log_sigma, accum);
  finalize_kernel<<<1, 1, 0, stream>>>(accum, log_sigma, out);

  (void)in_sizes; (void)n_in; (void)out_size; (void)ws_size;
}
